// Attention_2946347565540
// MI455X (gfx1250) — matrix operand order repaired and checked by reading, not compiled, not run on hardware
//
#include <hip/hip_runtime.h>

// ---------------------------------------------------------------------------
// Problem constants (from reference): B=2, L=2048, D=4096, H=32, KVH=8, HD=128
// ---------------------------------------------------------------------------
#define B_SZ   2
#define L_SEQ  2048
#define D_MOD  4096
#define NH     32
#define NKV    8
#define HD     128
#define REP    4            // NH / NKV
#define QKV_N  6144         // 4096 q + 1024 k + 1024 v
#define ROWS   (B_SZ * L_SEQ)   // 4096

typedef __attribute__((ext_vector_type(16))) __bf16 v16bf;
typedef __attribute__((ext_vector_type(8)))  float  v8f;

union AFrag { v16bf v; unsigned u[8]; };
union BFrag { v16bf v; unsigned u[8]; };

static __device__ __forceinline__ unsigned short f2bf(float f) {
    union { float f; unsigned u; } x; x.f = f;
    unsigned r = x.u + 0x7FFFu + ((x.u >> 16) & 1u);   // round-to-nearest-even
    return (unsigned short)(r >> 16);
}
static __device__ __forceinline__ float bf2f(unsigned short b) {
    union { unsigned u; float f; } x; x.u = ((unsigned)b) << 16;
    return x.f;
}

// ---------------------------------------------------------------------------
// CDNA5 async global->LDS copy (ASYNCcnt-tracked DMA), with sync fallback
// ---------------------------------------------------------------------------
#define GLOBAL_AS __attribute__((address_space(1)))
#define LDS_AS    __attribute__((address_space(3)))

// 16-byte payload type matching the builtin's expected pointee
// (diagnostic showed: __attribute__((__vector_size__(4*sizeof(int)))) int)
typedef int b128_t __attribute__((vector_size(16)));

#if __has_builtin(__builtin_amdgcn_global_load_async_to_lds_b128) && \
    __has_builtin(__builtin_amdgcn_s_wait_asynccnt)
#define HAVE_ASYNC 1
// copy 16 bytes global -> LDS without touching VGPR data path
#define ASYNC_CP16(gsrc, ldst)                                               \
    __builtin_amdgcn_global_load_async_to_lds_b128(                          \
        (GLOBAL_AS b128_t*)(gsrc), (LDS_AS b128_t*)(ldst), 0, 0)
#define ASYNC_WAIT(n) __builtin_amdgcn_s_wait_asynccnt(n)
#else
#define HAVE_ASYNC 0
static __device__ __forceinline__ void sync_cp16(const void* g, void* l) {
    *(uint4*)l = *(const uint4*)g;
}
#define ASYNC_CP16(gsrc, ldst) sync_cp16((const void*)(gsrc), (void*)(ldst))
#define ASYNC_WAIT(n) do { } while (0)
#endif

// ---------------------------------------------------------------------------
// f32 -> bf16 conversion
// ---------------------------------------------------------------------------
__global__ void cvt_f32_bf16(const float* __restrict__ src,
                             unsigned short* __restrict__ dst, int n) {
    int i = blockIdx.x * blockDim.x + threadIdx.x;
    if (i < n) dst[i] = f2bf(src[i]);
}

// Pack [wq | wk | wv] into one bf16 matrix of shape [D_MOD][QKV_N]
__global__ void pack_wqkv(const float* __restrict__ wq,
                          const float* __restrict__ wk,
                          const float* __restrict__ wv,
                          unsigned short* __restrict__ dst) {
    int t = blockIdx.x * blockDim.x + threadIdx.x;
    if (t >= D_MOD * QKV_N) return;
    int n = t % QKV_N;
    int k = t / QKV_N;
    float val;
    if (n < 4096)        val = wq[(size_t)k * 4096 + n];
    else if (n < 5120)   val = wk[(size_t)k * 1024 + (n - 4096)];
    else                 val = wv[(size_t)k * 1024 + (n - 5120)];
    dst[t] = f2bf(val);
}

// ---------------------------------------------------------------------------
// Tiled bf16 WMMA GEMM:  C[M,N] = A[M,K] * B[K,N]
// Block = 256 threads (8 waves), tile 128x128, K-step 32, double-buffered LDS
// staged via async global->LDS DMA.  Each wave: 4x2 tiles of 16x16 WMMA.
// ---------------------------------------------------------------------------
#define BM 128
#define BN 128
#define BK 32

template <bool OUT_BF16>
__global__ __launch_bounds__(256) void gemm_bf16(
    const unsigned short* __restrict__ A,
    const unsigned short* __restrict__ B,
    void* __restrict__ C,
    int M, int N, int K)
{
    __shared__ unsigned short As[2][BM][BK + 8];    // stride 40 ushort = 80 B
    __shared__ unsigned short Bs[2][BK][BN + 8];    // stride 136 ushort = 272 B

    const int tid   = threadIdx.x;
    const int lane  = tid & 31;
    const int w     = tid >> 5;        // 0..7
    const int wm    = w >> 2;          // 0..1
    const int wn    = w & 3;           // 0..3
    const int l15   = lane & 15;
    const int lhalf = lane >> 4;

    const int mBase = blockIdx.y * BM;
    const int nBase = blockIdx.x * BN;
    const int mWave = mBase + wm * 64;
    const int nWave = nBase + wn * 32;

    // staging coordinates (per thread: 32 B of A, 32 B of B per K-step)
    const int ar = tid >> 1, ac = (tid & 1) * 16;
    const int br = tid >> 3, bc = (tid & 7) * 16;

    v8f acc[4][2];
#pragma unroll
    for (int i = 0; i < 4; ++i)
#pragma unroll
        for (int j = 0; j < 2; ++j) acc[i][j] = {};

    const int kb = lhalf * 16;     // A-frag K base for this lane half
    const int kh = lhalf * 16;     // B-frag K base for this lane half

    const int nk = K / BK;

    // issue async copies for K-step kt into LDS buffer `buf` (4 x b128/thread)
    auto issue = [&](int buf, int k0) {
        const unsigned short* as = A + (size_t)(mBase + ar) * K + k0 + ac;
        ASYNC_CP16(as,     &As[buf][ar][ac]);
        ASYNC_CP16(as + 8, &As[buf][ar][ac + 8]);
        const unsigned short* bs = B + (size_t)(k0 + br) * N + nBase + bc;
        ASYNC_CP16(bs,     &Bs[buf][br][bc]);
        ASYNC_CP16(bs + 8, &Bs[buf][br][bc + 8]);
    };

    issue(0, 0);

    for (int kt = 0; kt < nk; ++kt) {
        const int buf = kt & 1;
        if (kt + 1 < nk) {
            issue(buf ^ 1, (kt + 1) * BK);   // prefetch next tile (DMA)
            ASYNC_WAIT(4);                   // current tile's 4 copies done
        } else {
            ASYNC_WAIT(0);
        }
        __syncthreads();

        // ---- build fragments from LDS ----
        AFrag af[4];
#pragma unroll
        for (int mt = 0; mt < 4; ++mt) {
            const unsigned short* row = &As[buf][wm * 64 + mt * 16 + l15][0];
#pragma unroll
            for (int v = 0; v < 8; ++v) {
                int kk = kb + 2 * v;
                af[mt].u[v] = *(const unsigned*)(row + kk);
            }
        }
        BFrag bfr[2];
#pragma unroll
        for (int nt = 0; nt < 2; ++nt) {
            int n = wn * 32 + nt * 16 + l15;
#pragma unroll
            for (int v = 0; v < 8; ++v) {
                unsigned lo = Bs[buf][kh + 2 * v][n];
                unsigned hi = Bs[buf][kh + 2 * v + 1][n];
                bfr[nt].u[v] = lo | (hi << 16);
            }
        }
        // ---- 8 WMMAs ----
#pragma unroll
        for (int mt = 0; mt < 4; ++mt)
#pragma unroll
            for (int nt = 0; nt < 2; ++nt)
                acc[mt][nt] = __builtin_amdgcn_wmma_f32_16x16x32_bf16(
                    false, af[mt].v, false, bfr[nt].v,
                    (short)0, acc[mt][nt], false, false);

        __syncthreads();   // all waves done reading `buf` before it is refilled
    }

    // ---- epilogue ----
#pragma unroll
    for (int mt = 0; mt < 4; ++mt) {
#pragma unroll
        for (int nt = 0; nt < 2; ++nt) {
            int col = nWave + nt * 16 + l15;
#pragma unroll
            for (int v = 0; v < 8; ++v) {
                int row = mWave + mt * 16 + v + 8 * lhalf;
                float val = acc[mt][nt][v];
                if constexpr (OUT_BF16)
                    ((unsigned short*)C)[(size_t)row * N + col] = f2bf(val);
                else
                    ((float*)C)[(size_t)row * N + col] = val;
            }
        }
    }
}

// ---------------------------------------------------------------------------
// RoPE in place on bf16 QKV buffer (q heads 0..31, k heads 32..39)
// ---------------------------------------------------------------------------
__global__ void rope_kernel(unsigned short* __restrict__ qkv,
                            const float* __restrict__ fcos,
                            const float* __restrict__ fsin) {
    const int PAIRS = HD / 2;         // 64
    const int SLOTS = NH + NKV;       // 40
    int t = blockIdx.x * blockDim.x + threadIdx.x;
    if (t >= ROWS * SLOTS * PAIRS) return;
    int pair = t % PAIRS;
    int rest = t / PAIRS;
    int slot = rest % SLOTS;
    int row  = rest / SLOTS;
    int pos  = row % L_SEQ;
    int col  = (slot < NH) ? (slot * HD + 2 * pair)
                           : (4096 + (slot - NH) * HD + 2 * pair);
    unsigned short* p = qkv + (size_t)row * QKV_N + col;
    float xr = bf2f(p[0]);
    float xi = bf2f(p[1]);
    float c = fcos[pos * PAIRS + pair];
    float s = fsin[pos * PAIRS + pair];
    p[0] = f2bf(xr * c - xi * s);
    p[1] = f2bf(xr * s + xi * c);
}

// ---------------------------------------------------------------------------
// Flash attention (causal, GQA).  1 block = 4 waves = 64 query rows of (b,h).
// K/V tiles (32 keys x 128) double-buffered in LDS via async DMA.
// Per 32-key step: 8 WMMA (Q.K^T) + online softmax + 8 WMMA (P.V).
// ---------------------------------------------------------------------------
__global__ __launch_bounds__(128) void attn_kernel(
    const unsigned short* __restrict__ qkv,    // [ROWS][QKV_N] bf16, RoPE'd
    unsigned short* __restrict__ attnout)      // [ROWS][NH*HD] bf16
{
    __shared__ unsigned short Ks[2][32][HD + 8];   // stride 136
    __shared__ unsigned short Vs[2][32][HD + 8];
    __shared__ unsigned short Ps[4][16][36];       // per-wave P transpose scratch

    const int tid   = threadIdx.x;
    const int lane  = tid & 31;
    const int w     = tid >> 5;
    const int l15   = lane & 15;
    const int lhalf = lane >> 4;

    const int nQB = L_SEQ / 64;                   // 32
    const int bh  = blockIdx.x / nQB;
    const int qb  = blockIdx.x % nQB;
    const int b   = bh / NH;
    const int h   = bh % NH;
    const int hk  = h / REP;
    const int q0  = qb * 64 + w * 16;             // wave's query-tile start

    const float scl = 0.08838834764831845f;       // 1/sqrt(128)

    // ---- load Q fragments (16 rows x 128, 4 chunks of K=32) ----
    AFrag qf[4];
    {
        const unsigned short* qrow =
            qkv + (size_t)(b * L_SEQ + q0 + l15) * QKV_N + h * HD;
        int kq = lhalf * 16;
#pragma unroll
        for (int c = 0; c < 4; ++c)
#pragma unroll
            for (int v = 0; v < 8; ++v) {
                int kk = kq + 2 * v;
                qf[c].u[v] = *(const unsigned*)(qrow + c * 32 + kk);
            }
    }

    v8f o[8];
#pragma unroll
    for (int g = 0; g < 8; ++g) o[g] = {};
    float mrow[8], lrow[8];
#pragma unroll
    for (int v = 0; v < 8; ++v) { mrow[v] = -3.0e38f; lrow[v] = 0.0f; }

    const int kh16 = lhalf * 16;
    const int nsteps = qb * 2 + 2;                // keys 0 .. qb*64+63

    // staging coordinates: thread -> (key, 64B span of head dim)
    const int skey = tid >> 2;
    const int sd0  = (tid & 3) * 32;

    // issue async K/V copies for step s into buffer `buf` (8 x b128/thread)
    auto issue = [&](int buf, int s) {
        const size_t rbase = (size_t)(b * L_SEQ + s * 32 + skey) * QKV_N;
        const unsigned short* ksrc = qkv + rbase + 4096 + hk * HD + sd0;
        const unsigned short* vsrc = qkv + rbase + 5120 + hk * HD + sd0;
#pragma unroll
        for (int i = 0; i < 4; ++i)
            ASYNC_CP16(ksrc + i * 8, &Ks[buf][skey][sd0] + i * 8);
#pragma unroll
        for (int i = 0; i < 4; ++i)
            ASYNC_CP16(vsrc + i * 8, &Vs[buf][skey][sd0] + i * 8);
    };

    issue(0, 0);

    for (int s = 0; s < nsteps; ++s) {
        const int j0  = s * 32;
        const int buf = s & 1;
        if (s + 1 < nsteps) {
            issue(buf ^ 1, s + 1);     // prefetch next K/V tile (DMA)
            ASYNC_WAIT(8);             // current tile's 8 copies done
        } else {
            ASYNC_WAIT(0);
        }
        __syncthreads();

        // ---- scores S = scale * Q.K^T, with causal mask ----
        float st[2][8];
#pragma unroll
        for (int nt = 0; nt < 2; ++nt) {
            v8f sacc = {};
            int key = nt * 16 + l15;
#pragma unroll
            for (int c = 0; c < 4; ++c) {
                BFrag kf;
#pragma unroll
                for (int v = 0; v < 8; ++v) {
                    int d = c * 32 + kh16 + 2 * v;
                    kf.u[v] = *(const unsigned*)&Ks[buf][key][d];
                }
                sacc = __builtin_amdgcn_wmma_f32_16x16x32_bf16(
                    false, qf[c].v, false, kf.v, (short)0, sacc, false, false);
            }
            int j = j0 + nt * 16 + l15;
#pragma unroll
            for (int v = 0; v < 8; ++v) {
                int m = q0 + v + 8 * lhalf;
                float x = sacc[v] * scl;
                st[nt][v] = (j <= m) ? x : -1.0e30f;
            }
        }

        // ---- online softmax (row reductions across 16-lane groups) ----
#pragma unroll
        for (int v = 0; v < 8; ++v) {
            float cm = fmaxf(st[0][v], st[1][v]);
#pragma unroll
            for (int msk = 1; msk <= 8; msk <<= 1)
                cm = fmaxf(cm, __shfl_xor(cm, msk, 32));
            float mnew = fmaxf(mrow[v], cm);
            float rs   = __expf(mrow[v] - mnew);
            float p0   = __expf(st[0][v] - mnew);
            float p1   = __expf(st[1][v] - mnew);
            float ps   = p0 + p1;
#pragma unroll
            for (int msk = 1; msk <= 8; msk <<= 1)
                ps += __shfl_xor(ps, msk, 32);
            lrow[v] = lrow[v] * rs + ps;
            mrow[v] = mnew;
#pragma unroll
            for (int g = 0; g < 8; ++g) o[g][v] *= rs;
            int r = v + 8 * lhalf;
            Ps[w][r][l15]      = f2bf(p0);
            Ps[w][r][16 + l15] = f2bf(p1);
        }
        // all P writes are intra-wave; fence LDS with CDNA5 split counter
        asm volatile("s_wait_dscnt 0" ::: "memory");

        // ---- P (16x32) as A-fragment ----
        AFrag pf;
        {
            const unsigned short* prow = &Ps[w][l15][0];
            int kq = lhalf * 16;
#pragma unroll
            for (int v = 0; v < 8; ++v) {
                int kk = kq + 2 * v;
                pf.u[v] = *(const unsigned*)(prow + kk);
            }
        }
        // ---- O += P.V (8 WMMAs across 128 output dims) ----
#pragma unroll
        for (int g = 0; g < 8; ++g) {
            BFrag vf;
            int d = g * 16 + l15;
#pragma unroll
            for (int v = 0; v < 8; ++v) {
                unsigned lo = Vs[buf][kh16 + 2 * v][d];
                unsigned hi = Vs[buf][kh16 + 2 * v + 1][d];
                vf.u[v] = lo | (hi << 16);
            }
            o[g] = __builtin_amdgcn_wmma_f32_16x16x32_bf16(
                false, pf.v, false, vf.v, (short)0, o[g], false, false);
        }

        __syncthreads();   // all waves done reading `buf` before refill
    }

    // ---- finalize: divide by row sum, store bf16 ----
#pragma unroll
    for (int v = 0; v < 8; ++v) {
        float inv = 1.0f / lrow[v];
        int m = q0 + v + 8 * lhalf;
        unsigned short* orow = attnout + (size_t)(b * L_SEQ + m) * (NH * HD) + h * HD;
#pragma unroll
        for (int g = 0; g < 8; ++g)
            orow[g * 16 + l15] = f2bf(o[g][v] * inv);
    }
}

// ---------------------------------------------------------------------------
// Launch
// ---------------------------------------------------------------------------
extern "C" void kernel_launch(void* const* d_in, const int* in_sizes, int n_in,
                              void* d_out, int out_size, void* d_ws, size_t ws_size,
                              hipStream_t stream) {
    (void)in_sizes; (void)n_in; (void)out_size; (void)ws_size;
    const float* x    = (const float*)d_in[0];
    const float* wq   = (const float*)d_in[1];
    const float* wk   = (const float*)d_in[2];
    const float* wv   = (const float*)d_in[3];
    const float* wo   = (const float*)d_in[4];
    const float* fcos = (const float*)d_in[5];
    const float* fsin = (const float*)d_in[6];
    // d_in[7] = mask (causal mask applied analytically), d_in[8] = start_pos (0)
    float* out = (float*)d_out;

    char* ws = (char*)d_ws;
    unsigned short* xb    = (unsigned short*)(ws);                         // 32 MB
    unsigned short* wqkvb = (unsigned short*)(ws + 33554432ull);           // 48 MB
    unsigned short* wob   = (unsigned short*)(ws + 83886080ull);           // 32 MB
    unsigned short* qkvb  = (unsigned short*)(ws + 117440512ull);          // 48 MB
    unsigned short* attnb = (unsigned short*)(ws + 167772160ull);          // 32 MB

    // 1. convert inputs to bf16
    {
        int n = ROWS * D_MOD;
        cvt_f32_bf16<<<(n + 255) / 256, 256, 0, stream>>>(x, xb, n);
    }
    {
        int n = D_MOD * QKV_N;
        pack_wqkv<<<(n + 255) / 256, 256, 0, stream>>>(wq, wk, wv, wqkvb);
    }
    {
        int n = D_MOD * D_MOD;
        cvt_f32_bf16<<<(n + 255) / 256, 256, 0, stream>>>(wo, wob, n);
    }

    // 2. QKV projection: [4096,4096] x [4096,6144] -> bf16
    gemm_bf16<true><<<dim3(QKV_N / BN, ROWS / BM), 256, 0, stream>>>(
        xb, wqkvb, qkvb, ROWS, QKV_N, D_MOD);

    // 3. RoPE on q and k heads
    {
        int n = ROWS * (NH + NKV) * (HD / 2);
        rope_kernel<<<(n + 255) / 256, 256, 0, stream>>>(qkvb, fcos, fsin);
    }

    // 4. causal GQA flash attention
    attn_kernel<<<B_SZ * NH * (L_SEQ / 64), 128, 0, stream>>>(qkvb, attnb);

    // 5. output projection: [4096,4096] x [4096,4096] -> f32 out
    gemm_bf16<false><<<dim3(D_MOD / BN, ROWS / BM), 256, 0, stream>>>(
        attnb, wob, out, ROWS, D_MOD, D_MOD);
}
